// PhaseODEFunc_38577396253385
// MI455X (gfx1250) — matrix operand order repaired and checked by reading, not compiled, not run on hardware
//
#include <hip/hip_runtime.h>

// ---------------------------------------------------------------------------
// PhaseODE dphi kernel for MI455X (gfx1250, wave32, WMMA).
//
//   m        = Cphi @ Cxi^T + Sphi @ Sxi^T            (two f16 WMMA GEMMs)
//   w        = softmax(m / N)
//   coupling = sinphi * (W @ Cxi) - cosphi * (W @ Sxi) (two f16 WMMA GEMMs)
//   dphi     = coupling + A * sin(omega*t - phi)       (fused into GEMM-2)
// ---------------------------------------------------------------------------

typedef __attribute__((ext_vector_type(16))) _Float16 v16h;
typedef __attribute__((ext_vector_type(8)))  _Float16 v8h;
typedef __attribute__((ext_vector_type(8)))  float    v8f;

#define TWO_PI_F    6.28318530717958647692f
#define BETA_F      1.0f
#define K_COUP_F    1.0f
#define A_ANC_F     0.08f
#define OMEGA_ANC_F (TWO_PI_F * 200.0f)

constexpr int Bdim = 64;
constexpr int Pdim = 256;
constexpr int Ndim = 4096;

// -------------------------------------------------------------------- prep --
// cos/sin of xi in (P,N) layout (for GEMM-1 B operand, k=n contiguous) and
// transposed (N,P) layout (for GEMM-2 B operand, k=p contiguous).
__global__ void prep_xi_kernel(const float* __restrict__ xi,
                               _Float16* __restrict__ Cxi,
                               _Float16* __restrict__ Sxi,
                               _Float16* __restrict__ CxiT,
                               _Float16* __restrict__ SxiT) {
    int idx = blockIdx.x * blockDim.x + threadIdx.x;   // exact grid: P*N threads
    int p = idx / Ndim;
    int n = idx % Ndim;
    float x = xi[idx];
    float s = sinf(x);
    float c = cosf(x);
    Cxi[idx]           = (_Float16)c;
    Sxi[idx]           = (_Float16)s;
    CxiT[n * Pdim + p] = (_Float16)c;
    SxiT[n * Pdim + p] = (_Float16)s;
}

__global__ void prep_phi_kernel(const float* __restrict__ phi,
                                _Float16* __restrict__ Cphi,
                                _Float16* __restrict__ Sphi) {
    int idx = blockIdx.x * blockDim.x + threadIdx.x;   // exact grid: B*N threads
    float x = phi[idx];
    Cphi[idx] = (_Float16)cosf(x);
    Sphi[idx] = (_Float16)sinf(x);
}

// -------------------------------------------------------- fragment loaders --
// A 16x32 f16 fragment: lane(0..15) row M=lane, K=0..15;
// lanes 16..31 same rows, K=16..31.  Two 16B contiguous loads.
__device__ __forceinline__ v16h load_a_frag(const _Float16* __restrict__ base,
                                            int row, int ldk, int k0,
                                            int half /*lane>>4*/) {
    const _Float16* ap = base + row * ldk + k0 + half * 16;
    v8h lo = *(const v8h*)(ap);
    v8h hi = *(const v8h*)(ap + 8);
    v16h a;
#pragma unroll
    for (int j = 0; j < 8; ++j) { a[j] = lo[j]; a[8 + j] = hi[j]; }
    return a;
}

// B 32x16 f16 fragment: lanes 0..15 column N=lane, K=0..15 in order;
// lanes 16..31 column N=lane-16, K=16..31.  One 32B contiguous load,
// provided source row (indexed by the B column) holds K contiguously.
__device__ __forceinline__ v16h load_b_frag(const _Float16* __restrict__ base,
                                            int col_row, int ldk, int k0,
                                            int half /*lane>>4*/) {
    return *(const v16h*)(base + col_row * ldk + k0 + half * 16);
}

// ------------------------------------------------------------------ GEMM 1 --
// m[b,p] = sum_n Cphi[b,n]*Cxi[p,n] + Sphi[b,n]*Sxi[p,n]
// Tiles: (B/16) x (P/16) = 4 x 16 = 64 tiles; 1 wave per tile; 8 waves/block.
__global__ void gemm_m_kernel(const _Float16* __restrict__ Cphi,
                              const _Float16* __restrict__ Sphi,
                              const _Float16* __restrict__ Cxi,
                              const _Float16* __restrict__ Sxi,
                              float* __restrict__ mbuf) {
    const int lane = threadIdx.x & 31;
    const int wave = threadIdx.x >> 5;
    const int tile = blockIdx.x * 8 + wave;         // 0..63
    const int b0 = (tile >> 4) * 16;                // B tile
    const int p0 = (tile & 15) * 16;                // P tile
    const int half = lane >> 4;
    const int l16  = lane & 15;

    v8f acc = {};
#pragma unroll 2
    for (int k0 = 0; k0 < Ndim; k0 += 32) {
        v16h aC = load_a_frag(Cphi, b0 + l16, Ndim, k0, half);
        v16h aS = load_a_frag(Sphi, b0 + l16, Ndim, k0, half);
        // B[k=n][p] = Cxi[p][n]  -> row (p0+l16) of Cxi, contiguous in n
        v16h bC = load_b_frag(Cxi, p0 + l16, Ndim, k0, half);
        v16h bS = load_b_frag(Sxi, p0 + l16, Ndim, k0, half);
        acc = __builtin_amdgcn_wmma_f32_16x16x32_f16(false, aC, false, bC,
                                                     (short)0, acc, false, false);
        acc = __builtin_amdgcn_wmma_f32_16x16x32_f16(false, aS, false, bS,
                                                     (short)0, acc, false, false);
    }
    // C/D layout: VGPR r, lane -> M = r + 8*half, N = l16
#pragma unroll
    for (int r = 0; r < 8; ++r)
        mbuf[(b0 + r + half * 8) * Pdim + p0 + l16] = acc[r];
}

// ----------------------------------------------------------------- softmax --
// One block per b-row; 256 threads = one per p.
__global__ void softmax_kernel(const float* __restrict__ mbuf,
                               _Float16* __restrict__ Wbuf) {
    __shared__ float sh[256];
    const int b = blockIdx.x;
    const int p = threadIdx.x;
    const float scale = BETA_F / (float)Ndim;
    float v = mbuf[b * Pdim + p] * scale;
    sh[p] = v;
    __syncthreads();
#pragma unroll
    for (int s = 128; s > 0; s >>= 1) {
        if (p < s) sh[p] = fmaxf(sh[p], sh[p + s]);
        __syncthreads();
    }
    float vmax = sh[0];
    __syncthreads();
    float e = expf(v - vmax);
    sh[p] = e;
    __syncthreads();
#pragma unroll
    for (int s = 128; s > 0; s >>= 1) {
        if (p < s) sh[p] += sh[p + s];
        __syncthreads();
    }
    Wbuf[b * Pdim + p] = (_Float16)(e / sh[0]);
}

// ------------------------------------------------------------------ GEMM 2 --
// CWc[b,n] = sum_p w[b,p]*Cxi[p,n],  CWs likewise with Sxi; fused epilogue:
// dphi = K*(sin(phi)*CWc - cos(phi)*CWs) + A*sin(omega*t - phi)
// Tiles: (B/16) x (N/16) = 4 x 256 = 1024 tiles; 8 waves/block -> 128 blocks.
__global__ void gemm_coupling_kernel(const _Float16* __restrict__ Wbuf,
                                     const _Float16* __restrict__ CxiT,
                                     const _Float16* __restrict__ SxiT,
                                     const float* __restrict__ phi,
                                     const float* __restrict__ tptr,
                                     float* __restrict__ out) {
    const int lane = threadIdx.x & 31;
    const int wave = threadIdx.x >> 5;
    const int tile = blockIdx.x * 8 + wave;         // 0..1023
    const int b0 = (tile >> 8) * 16;                // B tile (tile / 256)
    const int n0 = (tile & 255) * 16;               // N tile
    const int half = lane >> 4;
    const int l16  = lane & 15;

    v8f accC = {};
    v8f accS = {};
#pragma unroll
    for (int k0 = 0; k0 < Pdim; k0 += 32) {
        v16h a  = load_a_frag(Wbuf, b0 + l16, Pdim, k0, half);
        // B[k=p][n] = CxiT[n][p] -> row (n0+l16) of CxiT, contiguous in p
        v16h bC = load_b_frag(CxiT, n0 + l16, Pdim, k0, half);
        v16h bS = load_b_frag(SxiT, n0 + l16, Pdim, k0, half);
        accC = __builtin_amdgcn_wmma_f32_16x16x32_f16(false, a, false, bC,
                                                      (short)0, accC, false, false);
        accS = __builtin_amdgcn_wmma_f32_16x16x32_f16(false, a, false, bS,
                                                      (short)0, accS, false, false);
    }
    const float t = tptr[0];
#pragma unroll
    for (int r = 0; r < 8; ++r) {
        int b = b0 + r + half * 8;
        int n = n0 + l16;
        float ph = phi[b * Ndim + n];
        float sp = sinf(ph);
        float cp = cosf(ph);
        float anc = A_ANC_F * sinf(OMEGA_ANC_F * t - ph);
        out[b * Ndim + n] = K_COUP_F * (sp * accC[r] - cp * accS[r]) + anc;
    }
}

// ------------------------------------------------------------------ launch --
extern "C" void kernel_launch(void* const* d_in, const int* in_sizes, int n_in,
                              void* d_out, int out_size, void* d_ws, size_t ws_size,
                              hipStream_t stream) {
    const float* t_dev  = (const float*)d_in[0];   // (1,)
    const float* phi    = (const float*)d_in[1];   // (B, N)
    const float* xi     = (const float*)d_in[2];   // (P, N)
    float*       out    = (float*)d_out;           // (B, N)

    char* ws = (char*)d_ws;
    const size_t MB = 1024u * 1024u;
    _Float16* Cxi  = (_Float16*)(ws + 0 * MB);           // P*N f16 = 2 MB
    _Float16* Sxi  = (_Float16*)(ws + 2 * MB);           // 2 MB
    _Float16* CxiT = (_Float16*)(ws + 4 * MB);           // 2 MB
    _Float16* SxiT = (_Float16*)(ws + 6 * MB);           // 2 MB
    _Float16* Cphi = (_Float16*)(ws + 8 * MB);           // B*N f16 = 512 KB
    _Float16* Sphi = (_Float16*)(ws + 8 * MB + 512u * 1024u);
    float*    mbuf = (float*)   (ws + 9 * MB);           // B*P f32 = 64 KB
    _Float16* Wbuf = (_Float16*)(ws + 9 * MB + 64u * 1024u);  // B*P f16 = 32 KB

    // 1) trig tables
    prep_xi_kernel<<<(Pdim * Ndim) / 256, 256, 0, stream>>>(xi, Cxi, Sxi, CxiT, SxiT);
    prep_phi_kernel<<<(Bdim * Ndim) / 256, 256, 0, stream>>>(phi, Cphi, Sphi);

    // 2) m = Cphi@Cxi^T + Sphi@Sxi^T  (64 tiles, 8 waves/block)
    gemm_m_kernel<<<8, 256, 0, stream>>>(Cphi, Sphi, Cxi, Sxi, mbuf);

    // 3) softmax over p per b-row
    softmax_kernel<<<Bdim, 256, 0, stream>>>(mbuf, Wbuf);

    // 4) coupling GEMMs + fused anchor epilogue (1024 tiles, 8 waves/block)
    gemm_coupling_kernel<<<128, 256, 0, stream>>>(Wbuf, CxiT, SxiT, phi, t_dev, out);
}
